// GaugeInvariantAttention_39625368273499
// MI455X (gfx1250) — matrix operand order repaired and checked by reading, not compiled, not run on hardware
//
#include <hip/hip_runtime.h>
#include <hip/hip_bf16.h>

// ---------------- CDNA5 WMMA types ----------------
typedef __bf16 bf16_t;
typedef bf16_t v16bf __attribute__((ext_vector_type(16)));
typedef float  v8f   __attribute__((ext_vector_type(8)));

union Frag16 { v16bf v; unsigned short s[16]; uint4 q[2]; };

__device__ __forceinline__ unsigned short f2bf(float x) {
  unsigned int u = __float_as_uint(x);
  u = (u + 0x7FFFu + ((u >> 16) & 1u)) >> 16;   // round-to-nearest-even
  return (unsigned short)u;
}
__device__ __forceinline__ unsigned int f2bf_pk(float lo, float hi) {
  return (unsigned int)f2bf(lo) | ((unsigned int)f2bf(hi) << 16);
}

// =====================================================================
// Generic tiled bf16 GEMM:  C[M,N](f32) = A[M,K](bf16) * op(B)(bf16)
//   bTrans==0 : B is [K,N] row-major       bTrans==1 : B is [N,K] row-major
// Assumes M%64==0, K%32==0 (true for every call site); only N is guarded,
// with loop-invariant predicates hoisted so the K loop is branch-free.
// block = 128 threads = 4 waves; block tile 64x64; wave tile 32x32.
// LDS tiles are [row][K] so all fragments are contiguous 16B LDS reads
// (ds_load_b128), per the ISA VGPR layouts.
// mode==1 ("scores") epilogue: val = (acc + hs*holo[b,m,n])*temp[h]*gate[b,n,h]
// =====================================================================
__global__ __launch_bounds__(128)
void gemm_bf16_kernel(const unsigned short* __restrict__ A,
                      const unsigned short* __restrict__ Bm,
                      float* __restrict__ C,
                      int M, int N, int K,
                      int lda, int ldb, int ldc,
                      long long aStride, long long bStride,
                      long long cOuter, long long cInner, int cInnerCnt,
                      int bTrans, int mode,
                      const float* __restrict__ holo,
                      const float* __restrict__ gate,
                      const float* __restrict__ temp,
                      float holoScale, int S, int H)
{
  __shared__ unsigned short As[64][40];   // [m][k]
  __shared__ unsigned short Bs[64][40];   // [n][k]

  const int bi = blockIdx.z;
  const unsigned short* Ab = A  + (long long)bi * aStride;
  const unsigned short* Bb = Bm + (long long)bi * bStride;
  float* Cb = C + (long long)(bi / cInnerCnt) * cOuter
                + (long long)(bi % cInnerCnt) * cInner;

  const int mBase = blockIdx.y * 64, nBase = blockIdx.x * 64;
  const int t = threadIdx.x, lane = t & 31, wave = t >> 5;
  const int wm = (wave >> 1) * 32, wn = (wave & 1) * 32;

  // per-thread tile-load coordinates
  const int lr  = t >> 3;          // 0..15  (rows, 4 apart by 16)
  const int lkc = (t & 7) << 2;    // 0,4,...,28
  const int lkb = t >> 4;          // 0..7   (B non-trans: k rows, +8 apart)
  const int lnc = (t & 15) << 2;   // 0,4,...,60

  // hoisted pointers (advance by 32 K-elements per step)
  const unsigned short* aP  = Ab + (long long)(mBase + lr) * lda + lkc;
  const long long aRow = (long long)16 * lda;
  const unsigned short* bPt = Bb + (long long)(nBase + lr) * ldb + lkc;  // bTrans
  const long long bRowT = (long long)16 * ldb;
  const unsigned short* bPn = Bb + (long long)lkb * ldb + nBase + lnc;   // !bTrans
  const long long bRowN = (long long)8 * ldb;
  const bool nOk = (nBase + lnc + 4) <= N;   // loop-invariant column guard

  v8f acc[2][2];
  #pragma unroll
  for (int i = 0; i < 2; ++i)
    #pragma unroll
    for (int j = 0; j < 2; ++j)
      #pragma unroll
      for (int r = 0; r < 8; ++r) acc[i][j][r] = 0.0f;

  for (int kk = 0; kk < K; kk += 32) {
    __syncthreads();
    // ---- A tile 64x32: contiguous K, raw 8B copies ----
    #pragma unroll
    for (int i = 0; i < 4; ++i) {
      const unsigned long long pv = *(const unsigned long long*)(aP + aRow * i);
      *(unsigned long long*)&As[lr + 16 * i][lkc] = pv;
    }
    // ---- B tile -> Bs[n][k] ----
    if (!bTrans) {
      #pragma unroll
      for (int i = 0; i < 4; ++i) {
        unsigned long long pv = *(const unsigned long long*)(bPn + bRowN * i);
        if (!nOk) pv = 0ull;                 // branch-free select
        const int kb = lkb + 8 * i;
        Bs[lnc + 0][kb] = (unsigned short)pv;
        Bs[lnc + 1][kb] = (unsigned short)(pv >> 16);
        Bs[lnc + 2][kb] = (unsigned short)(pv >> 32);
        Bs[lnc + 3][kb] = (unsigned short)(pv >> 48);
      }
      bPn += (long long)32 * ldb;
    } else {
      #pragma unroll
      for (int i = 0; i < 4; ++i) {
        const unsigned long long pv = *(const unsigned long long*)(bPt + bRowT * i);
        *(unsigned long long*)&Bs[lr + 16 * i][lkc] = pv;
      }
      bPt += 32;
    }
    aP += 32;
    __syncthreads();

    // ---- fragments (contiguous 16B LDS reads) + WMMA ----
    Frag16 af[2], bf2[2];
    const int mrow = lane & 15;
    const int ako  = (lane >= 16) ? 8  : 0;   // A: hi half-lanes start at K=8
    const int bko  = (lane >= 16) ? 8  : 0;   // B: hi half-lanes start at K=8
    #pragma unroll
    for (int fm = 0; fm < 2; ++fm) {
      const int r = wm + fm * 16 + mrow;
      af[fm].q[0] = *(const uint4*)&As[r][ako];
      af[fm].q[1] = *(const uint4*)&As[r][ako + 16];
    }
    #pragma unroll
    for (int fn = 0; fn < 2; ++fn) {
      const int c = wn + fn * 16 + mrow;
      bf2[fn].q[0] = *(const uint4*)&Bs[c][bko];
      bf2[fn].q[1] = *(const uint4*)&Bs[c][bko + 16];
    }
    #pragma unroll
    for (int fm = 0; fm < 2; ++fm)
      #pragma unroll
      for (int fn = 0; fn < 2; ++fn)
        acc[fm][fn] = __builtin_amdgcn_wmma_f32_16x16x32_bf16(
            false, af[fm].v, false, bf2[fn].v, (short)0, acc[fm][fn], false, false);
  }

  // ---- epilogue ----
  const int rowOff = (lane >= 16) ? 8 : 0;
  const int colOff = lane & 15;
  #pragma unroll
  for (int fm = 0; fm < 2; ++fm)
    #pragma unroll
    for (int fn = 0; fn < 2; ++fn) {
      const int n = nBase + wn + fn * 16 + colOff;
      if (n >= N) continue;
      #pragma unroll
      for (int r = 0; r < 8; ++r) {
        const int m = mBase + wm + fm * 16 + r + rowOff;
        float val = acc[fm][fn][r];
        if (mode == 1) {
          const int b = bi / H, h = bi % H;
          val = (val + holoScale * holo[(long long)b * S * S + (long long)m * S + n])
                * temp[h] * gate[((long long)b * S + n) * H + h];
        }
        Cb[(long long)m * ldc + n] = val;
      }
    }
}

// =====================================================================
// Gram/holonomy kernel: holo[b,i,j] = 2*(c_i . c_j)/F - sq_i - sq_j
// fp32 global float4 loads -> bf16 LDS (packed b64 stores) ->
// v_wmma_f32_16x16x32_bf16, K=F=147456.
// Double-buffered LDS ping-pong with register staging so the HBM stream
// (23.3 TB/s) overlaps WMMA issue. All tile pointers advance by constant
// strides; the K loop carries no address multiplies.
// =====================================================================
__global__ __launch_bounds__(128)
void gram_holo_kernel(const float* __restrict__ conn,
                      const float* __restrict__ sq,
                      float* __restrict__ holo, int S, int F)
{
  __shared__ unsigned short As[2][64][40];
  __shared__ unsigned short Bs[2][64][40];

  const int b = blockIdx.z;
  const float* Cb = conn + (long long)b * S * F;
  const int mBase = blockIdx.y * 64, nBase = blockIdx.x * 64;
  const int t = threadIdx.x, lane = t & 31, wave = t >> 5;
  const int wm = (wave >> 1) * 32, wn = (wave & 1) * 32;

  const int lr  = t >> 3;          // 0..15
  const int lkc = (t & 7) << 2;    // 0,4,...,28
  const long long rStep = (long long)16 * F;

  const float* ap[4];
  const float* bp[4];
  #pragma unroll
  for (int i = 0; i < 4; ++i) {
    ap[i] = Cb + (long long)(mBase + lr) * F + lkc + rStep * i;
    bp[i] = Cb + (long long)(nBase + lr) * F + lkc + rStep * i;
  }

  v8f acc[2][2];
  #pragma unroll
  for (int i = 0; i < 2; ++i)
    #pragma unroll
    for (int j = 0; j < 2; ++j)
      #pragma unroll
      for (int r = 0; r < 8; ++r) acc[i][j][r] = 0.0f;

  float4 aR[4], bR[4];
  // ---- prologue: stage tile 0 and store to buffer 0 ----
  #pragma unroll
  for (int i = 0; i < 4; ++i) {
    aR[i] = *(const float4*)ap[i];
    bR[i] = *(const float4*)bp[i];
    ap[i] += 32;
    bp[i] += 32;
  }
  #pragma unroll
  for (int i = 0; i < 4; ++i) {
    const int row = lr + 16 * i;
    *(uint2*)&As[0][row][lkc] = make_uint2(f2bf_pk(aR[i].x, aR[i].y), f2bf_pk(aR[i].z, aR[i].w));
    *(uint2*)&Bs[0][row][lkc] = make_uint2(f2bf_pk(bR[i].x, bR[i].y), f2bf_pk(bR[i].z, bR[i].w));
  }
  __syncthreads();

  const int nIter = F >> 5;   // F/32
  for (int it = 0; it < nIter; ++it) {
    const int cur = it & 1;
    const bool more = (it + 1) < nIter;
    // stage next tile while computing on current
    if (more) {
      #pragma unroll
      for (int i = 0; i < 4; ++i) {
        aR[i] = *(const float4*)ap[i];
        bR[i] = *(const float4*)bp[i];
        ap[i] += 32;
        bp[i] += 32;
      }
    }

    // ---- fragments + WMMA on current buffer ----
    Frag16 af[2], bf2[2];
    const int mrow = lane & 15;
    const int ako = (lane >= 16) ? 8 : 0;
    const int bko = (lane >= 16) ? 8 : 0;
    #pragma unroll
    for (int fm = 0; fm < 2; ++fm) {
      const int r = wm + fm * 16 + mrow;
      af[fm].q[0] = *(const uint4*)&As[cur][r][ako];
      af[fm].q[1] = *(const uint4*)&As[cur][r][ako + 16];
    }
    #pragma unroll
    for (int fn = 0; fn < 2; ++fn) {
      const int c = wn + fn * 16 + mrow;
      bf2[fn].q[0] = *(const uint4*)&Bs[cur][c][bko];
      bf2[fn].q[1] = *(const uint4*)&Bs[cur][c][bko + 16];
    }
    #pragma unroll
    for (int fm = 0; fm < 2; ++fm)
      #pragma unroll
      for (int fn = 0; fn < 2; ++fn)
        acc[fm][fn] = __builtin_amdgcn_wmma_f32_16x16x32_bf16(
            false, af[fm].v, false, bf2[fn].v, (short)0, acc[fm][fn], false, false);

    if (more) {
      const int nxt = 1 - cur;
      #pragma unroll
      for (int i = 0; i < 4; ++i) {
        const int row = lr + 16 * i;
        *(uint2*)&As[nxt][row][lkc] = make_uint2(f2bf_pk(aR[i].x, aR[i].y), f2bf_pk(aR[i].z, aR[i].w));
        *(uint2*)&Bs[nxt][row][lkc] = make_uint2(f2bf_pk(bR[i].x, bR[i].y), f2bf_pk(bR[i].z, bR[i].w));
      }
    }
    __syncthreads();
  }

  const float invF = 1.0f / (float)F;
  const int rowOff = (lane >= 16) ? 8 : 0;
  const int colOff = lane & 15;
  #pragma unroll
  for (int fm = 0; fm < 2; ++fm)
    #pragma unroll
    for (int fn = 0; fn < 2; ++fn)
      #pragma unroll
      for (int r = 0; r < 8; ++r) {
        const int m = mBase + wm + fm * 16 + r + rowOff;
        const int n = nBase + wn + fn * 16 + colOff;
        holo[(long long)b * S * S + (long long)m * S + n] =
            2.0f * acc[fm][fn][r] * invF - sq[b * S + m] - sq[b * S + n];
      }
}

// ------------------- small helper kernels -------------------
__global__ void f2bf_kernel(const float* __restrict__ src,
                            unsigned short* __restrict__ dst, long long n) {
  long long i = (long long)blockIdx.x * blockDim.x + threadIdx.x;
  if (i < n) dst[i] = f2bf(src[i]);
}

__global__ void sq_kernel(const float* __restrict__ conn, float* __restrict__ sq, int F) {
  __shared__ float red[256];
  const float* row = conn + (long long)blockIdx.x * F;
  float s = 0.0f;
  for (int i = threadIdx.x * 4; i < F; i += 256 * 4) {
    const float4 c = *(const float4*)(row + i);
    s += c.x * c.x + c.y * c.y + c.z * c.z + c.w * c.w;
  }
  red[threadIdx.x] = s; __syncthreads();
  for (int st = 128; st > 0; st >>= 1) {
    if (threadIdx.x < st) red[threadIdx.x] += red[threadIdx.x + st];
    __syncthreads();
  }
  if (threadIdx.x == 0) sq[blockIdx.x] = red[0] / (float)F;
}

__global__ void gate_kernel(const float* __restrict__ curv, const float* __restrict__ gk,
                            float* __restrict__ gate, int BS, int D, int H) {
  const int idx = blockIdx.x * blockDim.x + threadIdx.x;
  if (idx >= BS * H) return;
  const int bs = idx / H, h = idx % H;
  const float* c = curv + (long long)bs * D;
  float s = 0.0f;
  for (int d = 0; d < D; ++d) s += c[d] * gk[d * H + h];
  gate[idx] = 1.0f / (1.0f + expf(-s));
}

__global__ void pack_qk_kernel(const float* __restrict__ src, unsigned short* __restrict__ dst,
                               int Bn, int Hn, int Sn, int KD, int KP, int D) {
  long long idx = (long long)blockIdx.x * blockDim.x + threadIdx.x;
  const long long total = (long long)Bn * Hn * Sn * KP;
  if (idx >= total) return;
  const int kd = (int)(idx % KP);
  long long r = idx / KP;
  const int s = (int)(r % Sn); r /= Sn;
  const int h = (int)(r % Hn);
  const int b = (int)(r / Hn);
  unsigned short v = 0;
  if (kd < KD) v = f2bf(src[((long long)(b * Sn + s)) * D + h * KD + kd]);
  dst[idx] = v;
}

__global__ void transport_kernel(const float* __restrict__ vsrc, const float* __restrict__ T,
                                 unsigned short* __restrict__ vt,
                                 int Bn, int Hn, int Sn, int HD, int D) {
  long long idx = (long long)blockIdx.x * blockDim.x + threadIdx.x;
  const long long total = (long long)Bn * Hn * Sn * HD;
  if (idx >= total) return;
  const int m = (int)(idx % HD);
  long long r = idx / HD;
  const int j = (int)(r % Sn); r /= Sn;
  const int h = (int)(r % Hn);
  const int b = (int)(r / Hn);
  const float* vrow = vsrc + ((long long)(b * Sn + j)) * D + h * HD;
  float s = 0.0f;
  for (int mp = 0; mp < HD; ++mp) s += vrow[mp] * T[mp * HD + m];
  vt[idx] = f2bf(s);
}

__global__ void softmax_kernel(const float* __restrict__ scores,
                               unsigned short* __restrict__ attn, int S) {
  __shared__ float red[128];
  const long long base = (long long)blockIdx.x * S;
  const int t = threadIdx.x;
  float mx = -3.4e38f;
  for (int j = t; j < S; j += 128) mx = fmaxf(mx, scores[base + j]);
  red[t] = mx; __syncthreads();
  for (int st = 64; st > 0; st >>= 1) {
    if (t < st) red[t] = fmaxf(red[t], red[t + st]);
    __syncthreads();
  }
  mx = red[0]; __syncthreads();
  float sm = 0.0f;
  for (int j = t; j < S; j += 128) sm += expf(scores[base + j] - mx);
  red[t] = sm; __syncthreads();
  for (int st = 64; st > 0; st >>= 1) {
    if (t < st) red[t] += red[t + st];
    __syncthreads();
  }
  const float inv = 1.0f / red[0];
  for (int j = t; j < S; j += 128)
    attn[base + j] = f2bf(expf(scores[base + j] - mx) * inv);
}

// =====================================================================
extern "C" void kernel_launch(void* const* d_in, const int* in_sizes, int n_in,
                              void* d_out, int out_size, void* d_ws, size_t ws_size,
                              hipStream_t stream)
{
  (void)in_sizes; (void)n_in; (void)out_size; (void)ws_size;

  const int Bn = 2, Sn = 512, Dn = 384, Hn = 8, KDn = 48, HDn = 48, KP = 64;
  const int Fn = Dn * Dn;        // 147456
  const int BSn = Bn * Sn;       // 1024
  const int NHn = Bn * Hn;       // 16
  const int HWn = Hn * KDn;      // 384

  const float* emb  = (const float*)d_in[0];
  const float* curv = (const float*)d_in[1];
  const float* conn = (const float*)d_in[2];
  const float* Wq   = (const float*)d_in[3];
  const float* Wk   = (const float*)d_in[4];
  const float* Wv   = (const float*)d_in[5];
  const float* Wo   = (const float*)d_in[6];
  const float* gk   = (const float*)d_in[7];
  const float* Tk   = (const float*)d_in[8];
  const float* temp = (const float*)d_in[9];

  char* p = (char*)d_ws;
  auto carve = [&](size_t bytes) -> void* {
    void* r = (void*)p;
    p += (bytes + 255) & ~(size_t)255;
    return r;
  };

  unsigned short* emb_bf = (unsigned short*)carve((size_t)BSn * Dn * 2);
  unsigned short* wq_bf  = (unsigned short*)carve((size_t)Dn * HWn * 2);
  unsigned short* wk_bf  = (unsigned short*)carve((size_t)Dn * HWn * 2);
  unsigned short* wv_bf  = (unsigned short*)carve((size_t)Dn * HWn * 2);
  unsigned short* wo_bf  = (unsigned short*)carve((size_t)HWn * Dn * 2);
  float* qf = (float*)carve((size_t)BSn * HWn * 4);
  float* kf = (float*)carve((size_t)BSn * HWn * 4);
  float* vf = (float*)carve((size_t)BSn * HWn * 4);
  unsigned short* q_bf  = (unsigned short*)carve((size_t)NHn * Sn * KP * 2);
  unsigned short* k_bf  = (unsigned short*)carve((size_t)NHn * Sn * KP * 2);
  unsigned short* vt_bf = (unsigned short*)carve((size_t)NHn * Sn * HDn * 2);
  float* sqb    = (float*)carve((size_t)BSn * 4);
  float* holo   = (float*)carve((size_t)Bn * Sn * Sn * 4);
  float* gate   = (float*)carve((size_t)BSn * Hn * 4);
  float* scores = (float*)carve((size_t)NHn * Sn * Sn * 4);
  unsigned short* attn_bf = (unsigned short*)carve((size_t)NHn * Sn * Sn * 2);
  float* ctxf = (float*)carve((size_t)BSn * HWn * 4);
  unsigned short* ctx_bf = (unsigned short*)carve((size_t)BSn * HWn * 2);

  // ---- 1. bf16 conversions of emb and weights ----
  {
    long long n = (long long)BSn * Dn;
    f2bf_kernel<<<(unsigned)((n + 255) / 256), 256, 0, stream>>>(emb, emb_bf, n);
    n = (long long)Dn * HWn;
    f2bf_kernel<<<(unsigned)((n + 255) / 256), 256, 0, stream>>>(Wq, wq_bf, n);
    f2bf_kernel<<<(unsigned)((n + 255) / 256), 256, 0, stream>>>(Wk, wk_bf, n);
    f2bf_kernel<<<(unsigned)((n + 255) / 256), 256, 0, stream>>>(Wv, wv_bf, n);
    f2bf_kernel<<<(unsigned)((n + 255) / 256), 256, 0, stream>>>(Wo, wo_bf, n);
  }

  // ---- 2. Q/K/V projections: [1024,384] x [384,384] ----
  {
    dim3 grid((HWn + 63) / 64, (BSn + 63) / 64, 1);
    gemm_bf16_kernel<<<grid, 128, 0, stream>>>(emb_bf, wq_bf, qf,
        BSn, HWn, Dn, Dn, HWn, HWn, 0LL, 0LL, 0LL, 0LL, 1, 0, 0,
        nullptr, nullptr, nullptr, 0.0f, Sn, Hn);
    gemm_bf16_kernel<<<grid, 128, 0, stream>>>(emb_bf, wk_bf, kf,
        BSn, HWn, Dn, Dn, HWn, HWn, 0LL, 0LL, 0LL, 0LL, 1, 0, 0,
        nullptr, nullptr, nullptr, 0.0f, Sn, Hn);
    gemm_bf16_kernel<<<grid, 128, 0, stream>>>(emb_bf, wv_bf, vf,
        BSn, HWn, Dn, Dn, HWn, HWn, 0LL, 0LL, 0LL, 0LL, 1, 0, 0,
        nullptr, nullptr, nullptr, 0.0f, Sn, Hn);
  }

  // ---- 3. sq[b,s] = ||conn_row||^2 / F ----
  sq_kernel<<<BSn, 256, 0, stream>>>(conn, sqb, Fn);

  // ---- 4. Gram + holonomy (the 155 GFLOP GEMM, K=147456) ----
  {
    dim3 grid(Sn / 64, Sn / 64, Bn);
    gram_holo_kernel<<<grid, 128, 0, stream>>>(conn, sqb, holo, Sn, Fn);
  }

  // ---- 5. curvature gate ----
  gate_kernel<<<(BSn * Hn + 255) / 256, 256, 0, stream>>>(curv, gk, gate, BSn, Dn, Hn);

  // ---- 6. pack q/k into [B,H,S,64] bf16 (zero-padded K), transport v ----
  {
    long long n = (long long)NHn * Sn * KP;
    pack_qk_kernel<<<(unsigned)((n + 255) / 256), 256, 0, stream>>>(qf, q_bf, Bn, Hn, Sn, KDn, KP, HWn);
    pack_qk_kernel<<<(unsigned)((n + 255) / 256), 256, 0, stream>>>(kf, k_bf, Bn, Hn, Sn, KDn, KP, HWn);
    n = (long long)NHn * Sn * HDn;
    transport_kernel<<<(unsigned)((n + 255) / 256), 256, 0, stream>>>(vf, Tk, vt_bf, Bn, Hn, Sn, HDn, HWn);
  }

  // ---- 7. scores = (q.k^T + 0.1*holo) * temp[h] * gate[b,j,h] ----
  {
    dim3 grid(Sn / 64, Sn / 64, NHn);
    gemm_bf16_kernel<<<grid, 128, 0, stream>>>(q_bf, k_bf, scores,
        Sn, Sn, KP, KP, KP, Sn,
        (long long)Sn * KP, (long long)Sn * KP,
        (long long)Sn * Sn, 0LL, 1,
        /*bTrans=*/1, /*mode=*/1, holo, gate, temp, 0.1f, Sn, Hn);
  }

  // ---- 8. softmax over keys -> bf16 attn ----
  softmax_kernel<<<NHn * Sn, 128, 0, stream>>>(scores, attn_bf, Sn);

  // ---- 9. ctx[b,h] = attn[b,h] (SxS) * v_t[b,h] (Sx48) -> ctx f32 [B*S,384] ----
  {
    dim3 grid((HDn + 63) / 64, Sn / 64, NHn);
    gemm_bf16_kernel<<<grid, 128, 0, stream>>>(attn_bf, vt_bf, ctxf,
        Sn, HDn, Sn, Sn, HDn, HWn,
        (long long)Sn * Sn, (long long)Sn * HDn,
        (long long)Sn * HWn, (long long)HDn, Hn,
        /*bTrans=*/0, /*mode=*/0, nullptr, nullptr, nullptr, 0.0f, Sn, Hn);
  }

  // ---- 10. out = ctx [1024,384] x Wo [384,384] ----
  {
    long long n = (long long)BSn * HWn;
    f2bf_kernel<<<(unsigned)((n + 255) / 256), 256, 0, stream>>>(ctxf, ctx_bf, n);
    dim3 grid((Dn + 63) / 64, (BSn + 63) / 64, 1);
    gemm_bf16_kernel<<<grid, 128, 0, stream>>>(ctx_bf, wo_bf, (float*)d_out,
        BSn, Dn, HWn, HWn, Dn, Dn, 0LL, 0LL, 0LL, 0LL, 1, 0, 0,
        nullptr, nullptr, nullptr, 0.0f, Sn, Hn);
  }
}
